// SAGE_3083786519241
// MI455X (gfx1250) — compile-verified, matrix operand order checked by reading, not run on hardware
//
#include <hip/hip_runtime.h>
#include <stdint.h>

// ---------------- CDNA5 WMMA types ----------------
typedef __attribute__((ext_vector_type(16))) __bf16 v16bf;
typedef __attribute__((ext_vector_type(8)))  float  v8f;

union FragU { uint32_t u[8]; v16bf v; };

__device__ __forceinline__ uint32_t f2bf(float x) {
    uint32_t u = __float_as_uint(x);
    return (u + 0x7FFFu + ((u >> 16) & 1u)) >> 16;   // RNE truncate to bf16
}
__device__ __forceinline__ uint32_t pack_bf(float lo, float hi) {
    return f2bf(lo) | (f2bf(hi) << 16);
}
__device__ __forceinline__ float sigm(float x) {
    return 1.0f / (1.0f + __expf(-x));
}
__device__ __forceinline__ float tanh_fast(float x) {
    return 2.0f * sigm(2.0f * x) - 1.0f;
}

__device__ __forceinline__ v8f wmma_bf16(v16bf a, v16bf b, v8f c) {
    // D = A(16x32 bf16) * B(32x16 bf16) + C(16x16 f32)
    return __builtin_amdgcn_wmma_f32_16x16x32_bf16(
        false, a, false, b, (short)0, c, false, false);
}

// Operand fragment from a bf16 matrix (LDS or global), row-major with leading dim ld.
// 16-bit operand layout (ISA 7.12.2): lane&15 = row (A: M, B: N col of W^T);
// lane>>4 picks K subset: kg=0 -> K = k0+{0..7,16..23}, kg=1 -> K = k0+{8..15,24..31}.
// u[0..3] and u[4..7] are each 16 contiguous bytes -> two b128 loads.
__device__ __forceinline__ v16bf load_frag_bf16(const uint16_t* __restrict__ base,
                                                int row, int ld, int k0, int kg) {
    FragU f;
    const uint16_t* r = base + (size_t)row * ld;
    const int kb = k0 + kg * 8;
#pragma unroll
    for (int v = 0; v < 4; ++v)
        f.u[v] = *reinterpret_cast<const uint32_t*>(r + kb + 2 * v);
#pragma unroll
    for (int v = 0; v < 4; ++v)
        f.u[4 + v] = *reinterpret_cast<const uint32_t*>(r + kb + 16 + 2 * v);
    return f.v;
}

// ---- one-time f32 -> bf16 conversion (feats and all weight matrices) ----
__global__ __launch_bounds__(256) void cvt_f32_to_bf16(const float* __restrict__ src,
                                                       uint16_t* __restrict__ dst, int n4) {
    const int i = blockIdx.x * blockDim.x + threadIdx.x;
    if (i < n4) {
        float4 p = ((const float4*)src)[i];
        ((uint2*)dst)[i] = make_uint2(pack_bf(p.x, p.y), pack_bf(p.z, p.w));
    }
}

// One SAGE layer with LSTM aggregation. 16 nodes per workgroup, 512 threads
// (16 wave32s). Wave w owns gate columns [32w, 32w+32): two 16x16 WMMA tiles,
// K = 256 = [x_t(128) | h_t(128)], bf16 weights resident in VGPRs for all 16 steps.
// Element group for the cell update: thread tid handles (m = wave, j = lane*4..+3).
template <int F_OUT, bool FINAL_SIG>
__global__ __launch_bounds__(512) void sage_lstm_layer(
    const uint16_t* __restrict__ x_in,    // [N][128] bf16
    const int*      __restrict__ nbr,     // [N][16]
    const uint16_t* __restrict__ w_ih,    // [512][128] bf16
    const uint16_t* __restrict__ w_hh,    // [512][128] bf16
    const float*    __restrict__ b_ih,    // [512]
    const float*    __restrict__ b_hh,    // [512]
    const uint16_t* __restrict__ w_self,  // [F_OUT][128] bf16
    const uint16_t* __restrict__ w_neigh, // [F_OUT][128] bf16
    const float*    __restrict__ b_out,   // [F_OUT]
    void* __restrict__ out)               // [N][F_OUT] bf16 (ws) or f32 (d_out)
{
    __shared__ __align__(16) uint16_t sA[16 * 256];   // A operand [x_t | h_t] bf16, 8 KB
    __shared__ __align__(16) float    sC[16 * 128];   // LSTM cell state, 8 KB
    __shared__ __align__(16) float    sG[16 * 512];   // raw gates i,f,g,o, 32 KB
    __shared__ float                  sB[512];        // b_ih + b_hh

    const int tid      = threadIdx.x;
    const int lane     = tid & 31;
    const int wave     = tid >> 5;
    const int n        = lane & 15;     // WMMA N-col / A M-row within tile
    const int kg       = lane >> 4;     // K-half selector
    const int nodeBase = blockIdx.x * 16;

    // ---- resident LSTM weight fragments: 2 tiles x 8 k-chunks = 128 VGPRs ----
    v16bf wf0[8], wf1[8];
#pragma unroll
    for (int kc = 0; kc < 8; ++kc) {
        const uint16_t* W = (kc < 4) ? w_ih : w_hh;   // K 0..127 = x, 128..255 = h
        const int k0 = (kc & 3) * 32;
        wf0[kc] = load_frag_bf16(W, wave * 32 + n,      128, k0, kg);
        wf1[kc] = load_frag_bf16(W, wave * 32 + 16 + n, 128, k0, kg);
    }

    // combined gate bias (512 threads == 512 gates)
    sB[tid] = b_ih[tid] + b_hh[tid];

    // LSTM cell: c' = sig(f)*c + sig(i)*tanh(g); h = sig(o)*tanh(c')
    auto cell = [](float gi, float gf, float gg, float go, float cp, float& h) -> float {
        float c = sigm(gf) * cp + sigm(gi) * tanh_fast(gg);
        h = sigm(go) * tanh_fast(c);
        return c;
    };

    // init: c = 0, h = 0, x_0 = gathered neighbor-0 features
    {
        const int m = wave, j = lane * 4;
        *(float4*)&sC[m * 128 + j] = make_float4(0.f, 0.f, 0.f, 0.f);
        *(uint32_t*)&sA[m * 256 + 128 + j]     = 0u;
        *(uint32_t*)&sA[m * 256 + 128 + j + 2] = 0u;
        const int src = nbr[(nodeBase + m) * 16 + 0];
        uint2 p = *(const uint2*)&x_in[(size_t)src * 128 + j];
        *(uint32_t*)&sA[m * 256 + j]     = p.x;
        *(uint32_t*)&sA[m * 256 + j + 2] = p.y;
    }
    __syncthreads();

#pragma unroll 1
    for (int step = 0; step < 16; ++step) {
        // ---- prefetch x_{t+1} (or self feats for the projection) into VGPRs:
        //      issued before the WMMA phase so HBM gather latency is hidden ----
        uint2 xr;
        {
            const int node = nodeBase + wave;
            const int src = (step < 15) ? nbr[node * 16 + step + 1] : node;
            xr = *(const uint2*)&x_in[(size_t)src * 128 + lane * 4];
        }

        // ---- gates = [x_t | h_t] @ [w_ih | w_hh]^T via WMMA ----
        v8f acc0 = {};
        v8f acc1 = {};
#pragma unroll
        for (int kc = 0; kc < 8; ++kc) {
            v16bf a = load_frag_bf16(sA, n, 256, kc * 32, kg);
            acc0 = wmma_bf16(a, wf0[kc], acc0);
            acc1 = wmma_bf16(a, wf1[kc], acc1);
        }
        const float bb0 = sB[wave * 32 + n];
        const float bb1 = sB[wave * 32 + 16 + n];
        // C/D layout: VGPR r -> (M = r + 8*kg, N = lane&15)
#pragma unroll
        for (int r = 0; r < 8; ++r) {
            const int m = r + 8 * kg;
            sG[m * 512 + wave * 32 + n]      = acc0[r] + bb0;
            sG[m * 512 + wave * 32 + 16 + n] = acc1[r] + bb1;
        }
        __syncthreads();

        // ---- elementwise LSTM cell update + commit prefetched x ----
        {
            const int m = wave, j = lane * 4;
            float4 GI = *(const float4*)&sG[m * 512 +   0 + j];
            float4 GF = *(const float4*)&sG[m * 512 + 128 + j];
            float4 GG = *(const float4*)&sG[m * 512 + 256 + j];
            float4 GO = *(const float4*)&sG[m * 512 + 384 + j];
            float4 C  = *(const float4*)&sC[m * 128 + j];
            float h0, h1, h2, h3;
            C.x = cell(GI.x, GF.x, GG.x, GO.x, C.x, h0);
            C.y = cell(GI.y, GF.y, GG.y, GO.y, C.y, h1);
            C.z = cell(GI.z, GF.z, GG.z, GO.z, C.z, h2);
            C.w = cell(GI.w, GF.w, GG.w, GO.w, C.w, h3);
            *(float4*)&sC[m * 128 + j] = C;
            *(uint32_t*)&sA[m * 256 + 128 + j]     = pack_bf(h0, h1);
            *(uint32_t*)&sA[m * 256 + 128 + j + 2] = pack_bf(h2, h3);
            *(uint32_t*)&sA[m * 256 + j]     = xr.x;
            *(uint32_t*)&sA[m * 256 + j + 2] = xr.y;
        }
        __syncthreads();
    }

    // ---- projection: out = [x_self | h_16] @ [w_self | w_neigh]^T + b ----
    if (wave < F_OUT / 16) {
        v8f acc = {};
#pragma unroll
        for (int kc = 0; kc < 8; ++kc) {
            v16bf a = load_frag_bf16(sA, n, 256, kc * 32, kg);
            const uint16_t* W = (kc < 4) ? w_self : w_neigh;
            const int k0 = (kc & 3) * 32;
            v16bf b = load_frag_bf16(W, wave * 16 + n, 128, k0, kg);
            acc = wmma_bf16(a, b, acc);
        }
        const int col = wave * 16 + n;
        const float bo = b_out[col];
#pragma unroll
        for (int r = 0; r < 8; ++r) {
            const int m = r + 8 * kg;
            float v = acc[r] + bo;
            if (FINAL_SIG) {
                ((float*)out)[(size_t)(nodeBase + m) * F_OUT + col] = sigm(v);
            } else {
                v = fmaxf(v, 0.0f);   // ReLU, store bf16 activations to workspace
                ((uint16_t*)out)[(size_t)(nodeBase + m) * F_OUT + col] = (uint16_t)f2bf(v);
            }
        }
    }
}

static inline void cvt(const float* src, uint16_t* dst, size_t count, hipStream_t s) {
    const int n4 = (int)(count / 4);
    cvt_f32_to_bf16<<<(n4 + 255) / 256, 256, 0, s>>>(src, dst, n4);
}

extern "C" void kernel_launch(void* const* d_in, const int* in_sizes, int n_in,
                              void* d_out, int out_size, void* d_ws, size_t ws_size,
                              hipStream_t stream) {
    const float* feats    = (const float*)d_in[0];
    const int*   nbr      = (const int*)  d_in[1];
    const float* w_ih1    = (const float*)d_in[2];
    const float* w_hh1    = (const float*)d_in[3];
    const float* b_ih1    = (const float*)d_in[4];
    const float* b_hh1    = (const float*)d_in[5];
    const float* w_self1  = (const float*)d_in[6];
    const float* w_neigh1 = (const float*)d_in[7];
    const float* b1       = (const float*)d_in[8];
    const float* w_ih2    = (const float*)d_in[9];
    const float* w_hh2    = (const float*)d_in[10];
    const float* b_ih2    = (const float*)d_in[11];
    const float* b_hh2    = (const float*)d_in[12];
    const float* w_self2  = (const float*)d_in[13];
    const float* w_neigh2 = (const float*)d_in[14];
    const float* b2       = (const float*)d_in[15];

    const int N = in_sizes[0] / 128;          // 50000

    // ---- workspace layout (bf16): feats | h1 | weight pool ----
    uint16_t* featsB   = (uint16_t*)d_ws;                 // N*128
    uint16_t* h1       = featsB + (size_t)N * 128;        // N*128
    uint16_t* w_ih1b   = h1 + (size_t)N * 128;            // 512*128
    uint16_t* w_hh1b   = w_ih1b   + 512 * 128;
    uint16_t* w_self1b = w_hh1b   + 512 * 128;            // 128*128
    uint16_t* w_neigh1b= w_self1b + 128 * 128;
    uint16_t* w_ih2b   = w_neigh1b+ 128 * 128;            // 512*128
    uint16_t* w_hh2b   = w_ih2b   + 512 * 128;
    uint16_t* w_self2b = w_hh2b   + 512 * 128;            // 64*128
    uint16_t* w_neigh2b= w_self2b + 64 * 128;

    // one-time conversions (deterministic; rerun every call)
    cvt(feats,    featsB,    (size_t)N * 128, stream);
    cvt(w_ih1,    w_ih1b,    512 * 128, stream);
    cvt(w_hh1,    w_hh1b,    512 * 128, stream);
    cvt(w_self1,  w_self1b,  128 * 128, stream);
    cvt(w_neigh1, w_neigh1b, 128 * 128, stream);
    cvt(w_ih2,    w_ih2b,    512 * 128, stream);
    cvt(w_hh2,    w_hh2b,    512 * 128, stream);
    cvt(w_self2,  w_self2b,  64 * 128, stream);
    cvt(w_neigh2, w_neigh2b, 64 * 128, stream);

    dim3 grid(N / 16), block(512);
    // Layer 1: bf16 feats in, bf16 ReLU activations out (workspace)
    sage_lstm_layer<128, false><<<grid, block, 0, stream>>>(
        featsB, nbr, w_ih1b, w_hh1b, b_ih1, b_hh1, w_self1b, w_neigh1b, b1, h1);
    // Layer 2: bf16 in, sigmoid f32 out
    sage_lstm_layer<64, true><<<grid, block, 0, stream>>>(
        h1, nbr, w_ih2b, w_hh2b, b_ih2, b_hh2, w_self2b, w_neigh2b, b2, d_out);
}
